// Str2Str_55430847922450
// MI455X (gfx1250) — compile-verified, matrix operand order checked by reading, not run on hardware
//
#include <hip/hip_runtime.h>
#include <hip/hip_bf16.h>
#include <math.h>

// ---------------- problem constants ----------------
#define LSEQ   512
#define C_Z    128
#define C_S    16
#define NH     12
#define C_HID  16
#define P_QK   4
#define P_V    8
#define C_RES  128
#define KE     192          // edge feature K padded: 128+36+1 = 165 -> 192
#define EFEAT  165
#define INF_V  100000.0f
#define EPS_V  1e-12f
#define CATD   2112         // H*(C_HID + 4*P_V + C_Z)

typedef _Float16 half_t;
typedef _Float16 h16 __attribute__((ext_vector_type(16)));
typedef float    f8  __attribute__((ext_vector_type(8)));
typedef unsigned int u32x4 __attribute__((ext_vector_type(4)));
typedef int i32x4v __attribute__((ext_vector_type(4)));
typedef int i32x8v __attribute__((ext_vector_type(8)));

#if defined(__has_builtin)
#if __has_builtin(__builtin_amdgcn_tensor_load_to_lds)
#define HAVE_TDM 1
#endif
#endif

// ---------------- WMMA helpers (CDNA5 16x16x32 f16, wave32) ----------------
// A (16xK slice, row-major, ld elements) and B (stored as N x K row-major, i.e.
// weights [out][in]) load with the identical per-lane pattern (ISA 7.12.2):
// lane L holds row (L&15), K-offsets {ko..ko+7, ko+16..ko+23}, ko = (L>>4)*8.
__device__ __forceinline__ h16 frag_ld(const half_t* base, int ld) {
  const int lane = threadIdx.x & 31;
  const half_t* p = base + (long)(lane & 15) * (long)ld + ((lane >> 4) << 3);
  h16 f;
#pragma unroll
  for (int e = 0; e < 8; ++e) { f[e] = p[e]; f[e + 8] = p[e + 16]; }
  return f;
}

__device__ __forceinline__ f8 wmma_f16(h16 a, h16 b, f8 c) {
  // emits v_wmma_f32_16x16x32_f16
  return __builtin_amdgcn_wmma_f32_16x16x32_f16(false, a, false, b, (short)0, c,
                                                false, false);
}

// D/C layout: lane l, vgpr r -> row = r + 8*(l>=16), col = l&15

// ---------------- TDM: async 2D tile load (global -> LDS), f16 elements ----
// D# per CDNA5 ISA: group0 = {count/flags, lds_addr, global_addr, type=2},
// group1 = {mask|data_size, tensor_dim0/1, tile_dim0/1/2, dim0_stride, ...}.
__device__ __forceinline__ void tdm_load_tile_f16(unsigned lds_off,
                                                  const void* gptr,
                                                  unsigned dim0, unsigned dim1,
                                                  unsigned stride0) {
#ifdef HAVE_TDM
  unsigned long long ga = (unsigned long long)gptr;
  u32x4 g0;
  g0[0] = 1u;                                            // count=1 (valid), no gather
  g0[1] = lds_off;                                       // lds_addr (bytes)
  g0[2] = (unsigned)ga;                                  // global_addr[31:0]
  g0[3] = ((unsigned)(ga >> 32) & 0x01ffffffu) | 0x80000000u;  // addr[56:32], type=2
  i32x8v g1;
  g1[0] = (int)(1u << 16);                               // data_size = 1 (2 bytes)
  g1[1] = (int)((dim0 & 0xffffu) << 16);                 // tensor_dim0[15:0]
  g1[2] = (int)((dim0 >> 16) | ((dim1 & 0xffffu) << 16));// td0[31:16] | td1[15:0]
  g1[3] = (int)((dim1 >> 16) | ((dim0 & 0xffffu) << 16));// td1[31:16] | tile_dim0
  g1[4] = (int)(dim1 & 0xffffu);                         // tile_dim1, tile_dim2=0
  g1[5] = (int)stride0;                                  // dim0_stride[31:0]
  g1[6] = 0;                                             // dim0_stride[47:32]
  g1[7] = 0;
  i32x4v gz = {0, 0, 0, 0};
#if __clang_major__ >= 23
  i32x8v gz8 = {0, 0, 0, 0, 0, 0, 0, 0};
  __builtin_amdgcn_tensor_load_to_lds(g0, g1, gz, gz, gz8, 0);
#else
  __builtin_amdgcn_tensor_load_to_lds(g0, g1, gz, gz, 0);
#endif
#endif
}

__device__ __forceinline__ void tdm_wait() {
#ifdef HAVE_TDM
  __builtin_amdgcn_s_wait_tensorcnt(0);
#endif
}

__device__ __forceinline__ float dot16(const float* w, const float* s) {
  float a = 0.f;
#pragma unroll
  for (int k = 0; k < 16; ++k) a += w[k] * s[k];
  return a;
}
__device__ __forceinline__ float dot128(const float* w, const float* s) {
  float a = 0.f;
  for (int k = 0; k < 128; ++k) a += w[k] * s[k];
  return a;
}
__device__ __forceinline__ void mm3(float* o, const float* A, const float* B) {
  for (int r = 0; r < 3; ++r)
    for (int c = 0; c < 3; ++c)
      o[r*3+c] = A[r*3]*B[c] + A[r*3+1]*B[3+c] + A[r*3+2]*B[6+c];
}
__device__ __forceinline__ void mv3(float* o, const float* A, const float* v) {
  for (int r = 0; r < 3; ++r) o[r] = A[r*3]*v[0] + A[r*3+1]*v[1] + A[r*3+2]*v[2];
}

// ---------------- kernel 1: repack embed_e weights (128 x 165) -> f16 128 x 192
__global__ void conv_we_kernel(const float* we, half_t* wWe) {
  int idx = blockIdx.x * blockDim.x + threadIdx.x;
  if (idx >= C_Z * KE) return;
  int n = idx / KE, k = idx - n * KE;
  wWe[idx] = (k < EFEAT) ? (half_t)we[n * EFEAT + k] : (half_t)0.f;
}

// ---------------- kernel 2: LN of input state (norm_s) -> state_initial
__global__ void state_ln_kernel(const float* st, const float* g, const float* b,
                                float* sn) {
  int i = blockIdx.x * blockDim.x + threadIdx.x;
  if (i >= LSEQ) return;
  const float* s = st + (size_t)i * C_S;
  float mu = 0.f;
  for (int c = 0; c < C_S; ++c) mu += s[c];
  mu *= (1.f / C_S);
  float v = 0.f;
  for (int c = 0; c < C_S; ++c) { float d = s[c] - mu; v += d * d; }
  float rs = rsqrtf(v * (1.f / C_S) + 1e-5f);
  for (int c = 0; c < C_S; ++c) sn[(size_t)i * C_S + c] = (s[c] - mu) * rs * g[c] + b[c];
}

// ---------------- kernel 3: IPA projections per residue ----------------
// qf/kf: [h][i][32] f16 (K padded 16->32, pad pre-zeroed)
// vcat : [h][48][j] f16  rows 0..15 = v channels, 16..39 = v_pts (p*3+x), 40..47 pad
// qpts/kpts: [i][h][p][3] f32 (rigid-transformed)
__global__ void ipa_proj_kernel(const float* sn, const float* rot, const float* trans,
                                const float* qw, const float* qb,
                                const float* kvw, const float* kvb,
                                const float* qpw, const float* qpb,
                                const float* kvpw, const float* kvpb,
                                half_t* qf, half_t* kf, half_t* vcat,
                                float* qpts, float* kpts) {
  int i = blockIdx.x, tid = threadIdx.x;
  __shared__ float s[C_S], Rm[9], Tv[3];
  if (tid < C_S) s[tid] = sn[(size_t)i * C_S + tid];
  if (tid >= 16 && tid < 25) Rm[tid - 16] = rot[(size_t)i * 9 + tid - 16];
  if (tid >= 32 && tid < 35) Tv[tid - 32] = trans[(size_t)i * 3 + tid - 32];
  __syncthreads();
  for (int t = tid; t < 768; t += blockDim.x) {
    if (t < 192) {                       // q
      int h = t >> 4, c = t & 15;
      float v = dot16(qw + (size_t)t * 16, s) + qb[t];
      qf[(size_t)(h * LSEQ + i) * 32 + c] = (half_t)v;
    } else if (t < 384) {                // k (kv rows h*32 + c)
      int u = t - 192, h = u >> 4, c = u & 15, row = h * 32 + c;
      float v = dot16(kvw + (size_t)row * 16, s) + kvb[row];
      kf[(size_t)(h * LSEQ + i) * 32 + c] = (half_t)v;
    } else if (t < 576) {                // v (kv rows h*32 + 16 + c), transposed
      int u = t - 384, h = u >> 4, c = u & 15, row = h * 32 + 16 + c;
      float v = dot16(kvw + (size_t)row * 16, s) + kvb[row];
      vcat[(size_t)(h * 48 + c) * LSEQ + i] = (half_t)v;
    } else if (t < 624) {                // q_pts, u = h*4+p
      int u = t - 576;
      float raw[3], g[3];
      for (int x = 0; x < 3; ++x) {
        int row = x * 48 + u;
        raw[x] = dot16(qpw + (size_t)row * 16, s) + qpb[row];
      }
      for (int x = 0; x < 3; ++x)
        g[x] = Rm[x*3]*raw[0] + Rm[x*3+1]*raw[1] + Rm[x*3+2]*raw[2] + Tv[x];
      for (int x = 0; x < 3; ++x)
        qpts[((size_t)i * NH * P_QK + u) * 3 + x] = g[x];
    } else {                             // kv_pts, u = h*12 + pp
      int u = t - 624, h = u / 12, pp = u - h * 12;
      float raw[3], g[3];
      for (int x = 0; x < 3; ++x) {
        int row = x * 96 + u;
        raw[x] = dot16(kvpw + (size_t)row * 16, s) + kvpb[row];
      }
      for (int x = 0; x < 3; ++x)
        g[x] = Rm[x*3]*raw[0] + Rm[x*3+1]*raw[1] + Rm[x*3+2]*raw[2] + Tv[x];
      if (pp < P_QK) {
        for (int x = 0; x < 3; ++x)
          kpts[((size_t)i * NH * P_QK + h * P_QK + pp) * 3 + x] = g[x];
      } else {
        int q = pp - P_QK;
        for (int x = 0; x < 3; ++x)
          vcat[(size_t)(h * 48 + 16 + q * 3 + x) * LSEQ + i] = (half_t)g[x];
      }
    }
  }
}

// ---------------- kernel 4: fused edge embedding (WMMA + TDM) ----------------
// 16 pairs (same i, consecutive j) per block, 256 thr = 8 waves, one 16-col tile each.
// TDM stages the 128x192 f16 weight tile into LDS asynchronously while the block
// does LN(z) -> concat(rbf, neighbor); then GEMM(192->128, f16 WMMA) -> +bias -> LN
// -> write z'T f16 ([i][c][j]) and head bias ([h][i][j]).
__global__ void edge_embed_kernel(const float* z, const float* rbf, const int* aatype,
                                  const float* gz, const float* bz,
                                  const half_t* wWe, const float* be,
                                  const float* ge, const float* beln,
                                  const float* bw, const float* bbias,
                                  half_t* zT, float* biasbuf) {
  __shared__ __align__(16) half_t wlds[C_Z * KE];   // 48 KB weight tile
  __shared__ __align__(16) half_t feat[16][KE];
  __shared__ float zt[16][C_Z];
  __shared__ float rmu[16], rrs[16];
  int tid = threadIdx.x;
  int pr0 = blockIdx.x * 16;
  int i = pr0 >> 9, j0 = pr0 & 511;
#ifdef HAVE_TDM
  if (tid < 32) {                        // one DMA per block, issued by wave 0
    tdm_load_tile_f16((unsigned)(size_t)&wlds[0], wWe, KE, C_Z, KE);
  }
#else
  for (int idx = tid; idx < C_Z * KE; idx += blockDim.x) wlds[idx] = wWe[idx];
#endif
  if (tid < 16) {                        // LN stats of z row
    const float* zp = z + ((size_t)(i * LSEQ + j0 + tid)) * C_Z;
    float mu = 0.f;
    for (int c = 0; c < C_Z; ++c) mu += zp[c];
    mu *= (1.f / C_Z);
    float v = 0.f;
    for (int c = 0; c < C_Z; ++c) { float d = zp[c] - mu; v += d * d; }
    rmu[tid] = mu; rrs[tid] = rsqrtf(v * (1.f / C_Z) + 1e-5f);
  }
  __syncthreads();
  for (int idx = tid; idx < 16 * KE; idx += blockDim.x) {
    int r = idx / KE, c = idx - r * KE, j = j0 + r;
    float v;
    if (c < C_Z) {
      v = (z[((size_t)(i * LSEQ + j)) * C_Z + c] - rmu[r]) * rrs[r] * gz[c] + bz[c];
    } else if (c < C_Z + 36) {
      v = rbf[((size_t)(i * LSEQ + j)) * 36 + (c - C_Z)];
    } else if (c == C_Z + 36) {
      float fs = (float)(aatype[j] - aatype[i]);
      float sg = (fs > 0.f ? 1.f : (fs < 0.f ? -1.f : 0.f));
      v = sg * logf(fabsf(fs) + 1.f);
    } else v = 0.f;
    feat[r][c] = (half_t)v;
  }
#ifdef HAVE_TDM
  if (tid < 32) tdm_wait();              // wave 0 drains TENSORcnt before barrier
#endif
  __syncthreads();
  {
    int wave = tid >> 5, lane = tid & 31, n0 = wave * 16;
    f8 acc = {0.f, 0.f, 0.f, 0.f, 0.f, 0.f, 0.f, 0.f};
#pragma unroll
    for (int kk = 0; kk < KE / 32; ++kk) {
      h16 a = frag_ld(&feat[0][0] + kk * 32, KE);
      h16 b = frag_ld(wlds + (size_t)n0 * KE + kk * 32, KE);
      acc = wmma_f16(a, b, acc);
    }
    int col = lane & 15, rb = (lane >> 4) << 3;
#pragma unroll
    for (int r = 0; r < 8; ++r) zt[r + rb][n0 + col] = acc[r] + be[n0 + col];
  }
  __syncthreads();
  if (tid < 16) {                        // LN stats (norm_edge)
    float mu = 0.f;
    for (int c = 0; c < C_Z; ++c) mu += zt[tid][c];
    mu *= (1.f / C_Z);
    float v = 0.f;
    for (int c = 0; c < C_Z; ++c) { float d = zt[tid][c] - mu; v += d * d; }
    rmu[tid] = mu; rrs[tid] = rsqrtf(v * (1.f / C_Z) + 1e-5f);
  }
  __syncthreads();
  for (int idx = tid; idx < 16 * C_Z; idx += blockDim.x) {
    int r = idx >> 7, c = idx & 127;
    float v = (zt[r][c] - rmu[r]) * rrs[r] * ge[c] + beln[c];
    zt[r][c] = v;  // each element owned by exactly one thread
    zT[((size_t)(i * C_Z + c) << 9) + (j0 + r)] = (half_t)v;
  }
  __syncthreads();
  for (int idx = tid; idx < 16 * NH; idx += blockDim.x) {   // b_w projection
    int r = idx / NH, h = idx - r * NH;
    float acc = bbias[h];
    for (int c = 0; c < C_Z; ++c) acc += zt[r][c] * bw[h * C_Z + c];
    biasbuf[((size_t)(h * LSEQ + i)) * LSEQ + (j0 + r)] = acc;
  }
}

// ---------------- kernel 5: attention logits + softmax ----------------
__global__ void attn_kernel(const half_t* qf, const half_t* kf,
                            const float* qpts, const float* kpts,
                            const float* biasbuf, const float* headw,
                            const float* mask, half_t* attn) {
  __shared__ float logit[16][LSEQ];      // 32 KB
  __shared__ float kl[LSEQ * 12];        // 24 KB: this head's k_pts
  __shared__ float ql[16 * 12];
  __shared__ float rmx[16], rsm[16];
  int h = blockIdx.x, i0 = blockIdx.y * 16, tid = threadIdx.x;
  int wave = tid >> 5, lane = tid & 31;
  const float sc_qk = 0.14433756729740643f;   // 1/sqrt(3*C_HID)
  // stage point clouds for this head into LDS
  for (int idx = tid; idx < LSEQ * 12; idx += blockDim.x) {
    int j = idx / 12, u = idx - j * 12;
    kl[idx] = kpts[((size_t)j * NH + h) * 12 + u];
  }
  if (tid < 16 * 12) {
    int r = tid / 12, u = tid - r * 12;
    ql[tid] = qpts[((size_t)(i0 + r) * NH + h) * 12 + u];
  }
  for (int jt = wave; jt < LSEQ / 16; jt += 8) {
    f8 acc = {0.f, 0.f, 0.f, 0.f, 0.f, 0.f, 0.f, 0.f};
    h16 a = frag_ld(qf + (size_t)(h * LSEQ + i0) * 32, 32);
    h16 b = frag_ld(kf + (size_t)(h * LSEQ + jt * 16) * 32, 32);
    acc = wmma_f16(a, b, acc);
    int col = lane & 15, rb = (lane >> 4) << 3;
#pragma unroll
    for (int r = 0; r < 8; ++r) logit[r + rb][jt * 16 + col] = acc[r] * sc_qk;
  }
  __syncthreads();
  float hw = log1pf(expf(headw[h])) * 0.13608276348795434f;  // sqrt(1/(3*18))
  const float sc_b = 0.5773502691896258f;                    // sqrt(1/3)
  for (int idx = tid; idx < 16 * LSEQ; idx += blockDim.x) {
    int r = idx >> 9, j = idx & 511, i = i0 + r;
    float d2 = 0.f;
#pragma unroll
    for (int p = 0; p < P_QK; ++p) {
      const float* qp = ql + r * 12 + p * 3;
      const float* kp = kl + j * 12 + p * 3;
      float dx = qp[0] - kp[0], dy = qp[1] - kp[1], dz = qp[2] - kp[2];
      d2 += dx * dx + dy * dy + dz * dz;
    }
    float v = logit[r][j] + sc_b * biasbuf[((size_t)(h * LSEQ + i)) * LSEQ + j]
            - 0.5f * hw * d2 + INF_V * (mask[i] * mask[j] - 1.f);
    logit[r][j] = v;
  }
  __syncthreads();
  if (tid < 16) {
    float m = -3.4e38f;
    for (int j = 0; j < LSEQ; ++j) m = fmaxf(m, logit[tid][j]);
    float s = 0.f;
    for (int j = 0; j < LSEQ; ++j) s += expf(logit[tid][j] - m);
    rmx[tid] = m; rsm[tid] = 1.f / s;
  }
  __syncthreads();
  for (int idx = tid; idx < 16 * LSEQ; idx += blockDim.x) {
    int r = idx >> 9, j = idx & 511;
    attn[((size_t)(h * LSEQ + i0 + r)) * LSEQ + j] =
        (half_t)(expf(logit[r][j] - rmx[r]) * rsm[r]);
  }
}

// ---------------- kernel 6: o and o_pt = attn @ [v | v_pts]  (WMMA) -------
__global__ void av_kernel(const half_t* attn, const half_t* vcat,
                          float* o, float* optraw) {
  int h = blockIdx.x, i0 = blockIdx.y * 16, tid = threadIdx.x;
  int wave = tid >> 5, lane = tid & 31, n0 = wave * 16;
  f8 acc = {0.f, 0.f, 0.f, 0.f, 0.f, 0.f, 0.f, 0.f};
  const half_t* Ab = attn + (size_t)(h * LSEQ + i0) * LSEQ;
  const half_t* Bb = vcat + (size_t)(h * 48 + n0) * LSEQ;
  for (int kk = 0; kk < LSEQ / 32; ++kk) {
    h16 a = frag_ld(Ab + kk * 32, LSEQ);
    h16 b = frag_ld(Bb + kk * 32, LSEQ);
    acc = wmma_f16(a, b, acc);
  }
  int col = lane & 15, rb = (lane >> 4) << 3;
#pragma unroll
  for (int r = 0; r < 8; ++r) {
    int i = i0 + r + rb, n = n0 + col;
    if (n < C_HID) o[((size_t)i * NH + h) * C_HID + n] = acc[r];
    else if (n < C_HID + P_V * 3)
      optraw[((size_t)i * NH + h) * (P_V * 3) + (n - C_HID)] = acc[r];
  }
}

// ---------------- kernel 7: o_pair = attn(h,:) @ z'(i)  (WMMA, batched over i)
__global__ void opair_kernel(const half_t* attn, const half_t* zT, float* opair) {
  int i = blockIdx.x, tid = threadIdx.x;
  int wave = tid >> 5, lane = tid & 31, n0 = wave * 16;
  f8 acc = {0.f, 0.f, 0.f, 0.f, 0.f, 0.f, 0.f, 0.f};
  const half_t* Ab = attn + (size_t)i * LSEQ;                 // rows h, stride L*L
  const half_t* Bb = zT + ((size_t)i * C_Z + n0) * LSEQ;      // rows c, stride L
  for (int kk = 0; kk < LSEQ / 32; ++kk) {
    __builtin_prefetch(Bb + kk * 32 + 256, 0, 0);
    h16 a = frag_ld(Ab + kk * 32, LSEQ * LSEQ);
    h16 b = frag_ld(Bb + kk * 32, LSEQ);
    acc = wmma_f16(a, b, acc);
  }
  int col = lane & 15, rb = (lane >> 4) << 3;
#pragma unroll
  for (int r = 0; r < 8; ++r) {
    int hh = r + rb, c = n0 + col;
    if (hh < NH) opair[((size_t)i * NH + hh) * C_Z + c] = acc[r];
  }
}

// ---------------- kernel 8: per-residue tail ----------------
struct FinalP {
  const float *sn, *o, *optraw, *opair, *rot, *trans;
  const float *out_w, *out_b, *lng, *lnb;
  const float *w1, *b1, *w2, *b2, *w3, *b3, *tlg, *tlb;
  const float *bbw, *bbb;
  const float *aiw, *aib, *a0w, *a0b;
  const float *r0w1, *r0b1, *r0w2, *r0b2, *r1w1, *r1b1, *r1w2, *r1b2;
  const float *aow, *aob;
  const float *dframes, *amask, *lpos;
  const int *aatype, *gidx;
  float *oxyz, *oR, *ot, *os;
};

__global__ void final_kernel(FinalP P) {
  __shared__ float cat[CATD];
  __shared__ float s0[16], s1[16], s2[16], s3[16], tb[16], ang[16];
  __shared__ float Rl[9], Tl[3], Rn[9], Tn[3];
  __shared__ float a[128], rel[128], tmp[128];
  __shared__ float alpha[8][2];
  __shared__ float fR[8][9], ft[8][3], Rg[8][9], tg[8][3];
  int i = blockIdx.x, tid = threadIdx.x;
  if (tid < 9) Rl[tid] = P.rot[(size_t)i * 9 + tid];
  if (tid < 3) Tl[tid] = P.trans[(size_t)i * 3 + tid];
  if (tid < 16) s0[tid] = P.sn[(size_t)i * 16 + tid];
  for (int idx = tid; idx < NH * C_HID; idx += blockDim.x)
    cat[idx] = P.o[(size_t)i * NH * C_HID + idx];
  for (int idx = tid; idx < NH * C_Z; idx += blockDim.x)
    cat[576 + idx] = P.opair[(size_t)i * NH * C_Z + idx];
  __syncthreads();
  if (tid < NH * P_V) {                  // rotate o_pt back: R^T (o_pt - t)
    int h = tid >> 3, p = tid & 7;
    const float* op = P.optraw + ((size_t)i * NH + h) * 24 + p * 3;
    float d0 = op[0] - Tl[0], d1 = op[1] - Tl[1], d2 = op[2] - Tl[2];
    float r0 = Rl[0]*d0 + Rl[3]*d1 + Rl[6]*d2;
    float r1 = Rl[1]*d0 + Rl[4]*d1 + Rl[7]*d2;
    float r2 = Rl[2]*d0 + Rl[5]*d1 + Rl[8]*d2;
    cat[192 + tid] = r0; cat[288 + tid] = r1; cat[384 + tid] = r2;
    cat[480 + tid] = sqrtf(r0*r0 + r1*r1 + r2*r2 + EPS_V);
  }
  __syncthreads();
  if (tid < 16) {                        // out projection + residual
    float acc = P.out_b[tid];
    const float* w = P.out_w + (size_t)tid * CATD;
    for (int k = 0; k < CATD; ++k) acc += w[k] * cat[k];
    s1[tid] = s0[tid] + acc;
  }
  __syncthreads();
  if (tid < 16) {                        // ln_ipa
    float mu = 0.f; for (int c = 0; c < 16; ++c) mu += s1[c];
    mu *= (1.f / 16);
    float v = 0.f; for (int c = 0; c < 16; ++c) { float d = s1[c]-mu; v += d*d; }
    s2[tid] = (s1[tid]-mu) * rsqrtf(v*(1.f/16)+1e-5f) * P.lng[tid] + P.lnb[tid];
  }
  __syncthreads();
  if (tid < 16) tb[tid] = fmaxf(dot16(P.w1 + tid*16, s2) + P.b1[tid], 0.f);
  __syncthreads();
  if (tid < 16) s1[tid] = fmaxf(dot16(P.w2 + tid*16, tb) + P.b2[tid], 0.f);
  __syncthreads();
  if (tid < 16) tb[tid] = dot16(P.w3 + tid*16, s1) + P.b3[tid] + s2[tid];
  __syncthreads();
  if (tid < 16) {                        // tr_ln -> output s
    float mu = 0.f; for (int c = 0; c < 16; ++c) mu += tb[c];
    mu *= (1.f / 16);
    float v = 0.f; for (int c = 0; c < 16; ++c) { float d = tb[c]-mu; v += d*d; }
    s3[tid] = (tb[tid]-mu) * rsqrtf(v*(1.f/16)+1e-5f) * P.tlg[tid] + P.tlb[tid];
    P.os[(size_t)i * 16 + tid] = s3[tid];
  }
  __syncthreads();
  if (tid == 0) {                        // backbone update
    float upd[6];
    for (int r = 0; r < 6; ++r) upd[r] = dot16(P.bbw + r*16, s3) + P.bbb[r];
    float q0 = 1.f, q1 = upd[0], q2 = upd[1], q3 = upd[2];
    float n = rsqrtf(q0*q0 + q1*q1 + q2*q2 + q3*q3);
    q0 *= n; q1 *= n; q2 *= n; q3 *= n;
    float Rq[9] = {
      q0*q0+q1*q1-q2*q2-q3*q3, 2*(q1*q2-q0*q3),         2*(q1*q3+q0*q2),
      2*(q1*q2+q0*q3),         q0*q0-q1*q1+q2*q2-q3*q3, 2*(q2*q3-q0*q1),
      2*(q1*q3-q0*q2),         2*(q2*q3+q0*q1),         q0*q0-q1*q1-q2*q2+q3*q3};
    mm3(Rn, Rl, Rq);
    for (int r = 0; r < 3; ++r)
      Tn[r] = Rl[r*3]*upd[3] + Rl[r*3+1]*upd[4] + Rl[r*3+2]*upd[5] + Tl[r];
    for (int k = 0; k < 9; ++k) P.oR[(size_t)i*9 + k] = Rn[k];
    for (int k = 0; k < 3; ++k) P.ot[(size_t)i*3 + k] = Tn[k];
  }
  __syncthreads();
  // ---- angle resnet ----
  if (tid < 16) { s1[tid] = fmaxf(s3[tid], 0.f); tb[tid] = fmaxf(s0[tid], 0.f); }
  __syncthreads();
  if (tid < 128)
    a[tid] = dot16(P.aiw + tid*16, s1) + P.aib[tid]
           + dot16(P.a0w + tid*16, tb) + P.a0b[tid];
  __syncthreads();
  for (int blk = 0; blk < 2; ++blk) {
    const float *bw1 = blk ? P.r1w1 : P.r0w1, *bb1 = blk ? P.r1b1 : P.r0b1;
    const float *bw2 = blk ? P.r1w2 : P.r0w2, *bb2 = blk ? P.r1b2 : P.r0b2;
    if (tid < 128) rel[tid] = fmaxf(a[tid], 0.f);
    __syncthreads();
    if (tid < 128) tmp[tid] = fmaxf(dot128(bw1 + (size_t)tid*128, rel) + bb1[tid], 0.f);
    __syncthreads();
    if (tid < 128) a[tid] += dot128(bw2 + (size_t)tid*128, tmp) + bb2[tid];
    __syncthreads();
  }
  if (tid < 128) rel[tid] = fmaxf(a[tid], 0.f);
  __syncthreads();
  if (tid < 14) ang[tid] = dot128(P.aow + (size_t)tid*128, rel) + P.aob[tid];
  __syncthreads();
  if (tid == 0) { alpha[0][0] = 0.f; alpha[0][1] = 1.f; }   // prepended bb angle
  if (tid < 7) {
    float sv = ang[tid*2], cv = ang[tid*2+1];
    float nn = sqrtf(fmaxf(sv*sv + cv*cv, EPS_V));
    alpha[tid+1][0] = sv / nn; alpha[tid+1][1] = cv / nn;
  }
  __syncthreads();
  int aa = P.aatype[i];
  if (tid < 8) {                         // per-group frame: dR @ rot(alpha)
    const float* d4 = P.dframes + ((size_t)aa * 8 + tid) * 16;
    float sv = alpha[tid][0], cv = alpha[tid][1];
    for (int r = 0; r < 3; ++r) {
      fR[tid][r*3+0] = d4[r*4+0];
      fR[tid][r*3+1] = d4[r*4+1]*cv + d4[r*4+2]*sv;
      fR[tid][r*3+2] = -d4[r*4+1]*sv + d4[r*4+2]*cv;
      ft[tid][r] = d4[r*4+3];
    }
  }
  __syncthreads();
  if (tid == 0) {                        // chain composition for groups 5..7
    float c2R[9], c2t[3], c3R[9], c3t[3], c4R[9], c4t[3], tv[3];
    mm3(c2R, fR[4], fR[5]); mv3(tv, fR[4], ft[5]);
    for (int x = 0; x < 3; ++x) c2t[x] = tv[x] + ft[4][x];
    mm3(c3R, c2R, fR[6]);   mv3(tv, c2R, ft[6]);
    for (int x = 0; x < 3; ++x) c3t[x] = tv[x] + c2t[x];
    mm3(c4R, c3R, fR[7]);   mv3(tv, c3R, ft[7]);
    for (int x = 0; x < 3; ++x) c4t[x] = tv[x] + c3t[x];
    for (int k = 0; k < 9; ++k) { fR[5][k] = c2R[k]; fR[6][k] = c3R[k]; fR[7][k] = c4R[k]; }
    for (int k = 0; k < 3; ++k) { ft[5][k] = c2t[k]; ft[6][k] = c3t[k]; ft[7][k] = c4t[k]; }
  }
  __syncthreads();
  if (tid < 8) {                         // to global frame (t scaled by 10)
    mm3(Rg[tid], Rn, fR[tid]);
    for (int r = 0; r < 3; ++r)
      tg[tid][r] = Rn[r*3]*ft[tid][0] + Rn[r*3+1]*ft[tid][1] + Rn[r*3+2]*ft[tid][2]
                 + 10.f * Tn[r];
  }
  __syncthreads();
  if (tid < 14) {                        // atom14
    int g = P.gidx[aa * 14 + tid];
    float m = P.amask[aa * 14 + tid];
    const float* lp = P.lpos + ((size_t)aa * 14 + tid) * 3;
    for (int x = 0; x < 3; ++x)
      P.oxyz[(size_t)i * 42 + tid * 3 + x] =
        (Rg[g][x*3]*lp[0] + Rg[g][x*3+1]*lp[1] + Rg[g][x*3+2]*lp[2] + tg[g][x]) * m;
  }
}

// ---------------- host launch ----------------
enum {
  I_M = 0, I_Z, I_STATE, I_RBF, I_AATYPE, I_ROT, I_TRANS, I_MASK, I_GIDX,
  P_NORM_M_G, P_NORM_M_B, P_NORM_Z_G, P_NORM_Z_B, P_NORM_S_G, P_NORM_S_B,
  P_EMBED_X_W, P_EMBED_X_B, P_NORM_NODE_G, P_NORM_NODE_B,
  P_EMBED_E_W, P_EMBED_E_B, P_NORM_EDGE_G, P_NORM_EDGE_B,
  P_Q_W, P_Q_B, P_KV_W, P_KV_B, P_QP_W, P_QP_B, P_KVP_W, P_KVP_B,
  P_B_W, P_B_B, P_HEAD_W, P_OUT_W, P_OUT_B, P_LN_IPA_G, P_LN_IPA_B,
  P_TR_W1, P_TR_B1, P_TR_W2, P_TR_B2, P_TR_W3, P_TR_B3, P_TR_LN_G, P_TR_LN_B,
  P_BB_W, P_BB_B, P_ANG_IN_W, P_ANG_IN_B, P_ANG_INIT_W, P_ANG_INIT_B,
  P_B0W1, P_B0B1, P_B0W2, P_B0B2, P_B1W1, P_B1B1, P_B1W2, P_B1B2,
  P_ANG_OUT_W, P_ANG_OUT_B, P_DEF_FRAMES, P_ATOM_MASK, P_LIT_POS
};

extern "C" void kernel_launch(void* const* d_in, const int* in_sizes, int n_in,
                              void* d_out, int out_size, void* d_ws, size_t ws_size,
                              hipStream_t stream) {
  (void)in_sizes; (void)n_in; (void)out_size; (void)ws_size;
  const float* z     = (const float*)d_in[I_Z];
  const float* state = (const float*)d_in[I_STATE];
  const float* rbf   = (const float*)d_in[I_RBF];
  const int*   aaty  = (const int*)d_in[I_AATYPE];
  const float* rot   = (const float*)d_in[I_ROT];
  const float* trans = (const float*)d_in[I_TRANS];
  const float* mask  = (const float*)d_in[I_MASK];
  const int*   gidx  = (const int*)d_in[I_GIDX];
  auto F = [&](int k) { return (const float*)d_in[k]; };

  // workspace layout
  size_t off = 0;
  auto alloc = [&](size_t bytes) {
    size_t p = off;
    off = (off + bytes + 255) & ~(size_t)255;
    return p;
  };
  char* wsb = (char*)d_ws;
  size_t o_zT   = alloc((size_t)LSEQ * C_Z * LSEQ * sizeof(half_t));   // z'T [i][c][j]
  size_t o_bias = alloc((size_t)NH * LSEQ * LSEQ * sizeof(float));     // [h][i][j]
  size_t o_attn = alloc((size_t)16 * LSEQ * LSEQ * sizeof(half_t));    // heads padded
  size_t o_qf   = alloc((size_t)NH * LSEQ * 32 * sizeof(half_t));
  size_t o_kf   = alloc((size_t)NH * LSEQ * 32 * sizeof(half_t));
  size_t o_vc   = alloc((size_t)NH * 48 * LSEQ * sizeof(half_t));
  size_t o_qp   = alloc((size_t)LSEQ * NH * P_QK * 3 * sizeof(float));
  size_t o_kp   = alloc((size_t)LSEQ * NH * P_QK * 3 * sizeof(float));
  size_t o_o    = alloc((size_t)LSEQ * NH * C_HID * sizeof(float));
  size_t o_opt  = alloc((size_t)LSEQ * NH * P_V * 3 * sizeof(float));
  size_t o_opr  = alloc((size_t)LSEQ * NH * C_Z * sizeof(float));
  size_t o_sn   = alloc((size_t)LSEQ * C_S * sizeof(float));
  size_t o_wWe  = alloc((size_t)C_Z * KE * sizeof(half_t));
  size_t total  = off;

  half_t* zT   = (half_t*)(wsb + o_zT);
  float*  bias = (float*)(wsb + o_bias);
  half_t* attn = (half_t*)(wsb + o_attn);
  half_t* qf   = (half_t*)(wsb + o_qf);
  half_t* kf   = (half_t*)(wsb + o_kf);
  half_t* vc   = (half_t*)(wsb + o_vc);
  float*  qp   = (float*)(wsb + o_qp);
  float*  kp   = (float*)(wsb + o_kp);
  float*  obuf = (float*)(wsb + o_o);
  float*  optr = (float*)(wsb + o_opt);
  float*  opr  = (float*)(wsb + o_opr);
  float*  sn   = (float*)(wsb + o_sn);
  half_t* wWe  = (half_t*)(wsb + o_wWe);

  // zero pads (qf/kf upper K, vcat pad rows, attn heads 12..15)
  hipMemsetAsync(d_ws, 0, total, stream);

  conv_we_kernel<<<(C_Z * KE + 255) / 256, 256, 0, stream>>>(F(P_EMBED_E_W), wWe);
  state_ln_kernel<<<(LSEQ + 127) / 128, 128, 0, stream>>>(
      state, F(P_NORM_S_G), F(P_NORM_S_B), sn);
  ipa_proj_kernel<<<LSEQ, 256, 0, stream>>>(
      sn, rot, trans, F(P_Q_W), F(P_Q_B), F(P_KV_W), F(P_KV_B),
      F(P_QP_W), F(P_QP_B), F(P_KVP_W), F(P_KVP_B), qf, kf, vc, qp, kp);
  edge_embed_kernel<<<(LSEQ * LSEQ) / 16, 256, 0, stream>>>(
      z, rbf, aaty, F(P_NORM_Z_G), F(P_NORM_Z_B), wWe, F(P_EMBED_E_B),
      F(P_NORM_EDGE_G), F(P_NORM_EDGE_B), F(P_B_W), F(P_B_B), zT, bias);
  attn_kernel<<<dim3(NH, LSEQ / 16), 256, 0, stream>>>(
      qf, kf, qp, kp, bias, F(P_HEAD_W), mask, attn);
  av_kernel<<<dim3(NH, LSEQ / 16), 96, 0, stream>>>(attn, vc, obuf, optr);
  opair_kernel<<<LSEQ, 256, 0, stream>>>(attn, zT, opr);

  float* out = (float*)d_out;
  FinalP fp;
  fp.sn = sn; fp.o = obuf; fp.optraw = optr; fp.opair = opr;
  fp.rot = rot; fp.trans = trans;
  fp.out_w = F(P_OUT_W); fp.out_b = F(P_OUT_B);
  fp.lng = F(P_LN_IPA_G); fp.lnb = F(P_LN_IPA_B);
  fp.w1 = F(P_TR_W1); fp.b1 = F(P_TR_B1);
  fp.w2 = F(P_TR_W2); fp.b2 = F(P_TR_B2);
  fp.w3 = F(P_TR_W3); fp.b3 = F(P_TR_B3);
  fp.tlg = F(P_TR_LN_G); fp.tlb = F(P_TR_LN_B);
  fp.bbw = F(P_BB_W); fp.bbb = F(P_BB_B);
  fp.aiw = F(P_ANG_IN_W); fp.aib = F(P_ANG_IN_B);
  fp.a0w = F(P_ANG_INIT_W); fp.a0b = F(P_ANG_INIT_B);
  fp.r0w1 = F(P_B0W1); fp.r0b1 = F(P_B0B1); fp.r0w2 = F(P_B0W2); fp.r0b2 = F(P_B0B2);
  fp.r1w1 = F(P_B1W1); fp.r1b1 = F(P_B1B1); fp.r1w2 = F(P_B1W2); fp.r1b2 = F(P_B1B2);
  fp.aow = F(P_ANG_OUT_W); fp.aob = F(P_ANG_OUT_B);
  fp.dframes = F(P_DEF_FRAMES); fp.amask = F(P_ATOM_MASK); fp.lpos = F(P_LIT_POS);
  fp.aatype = aaty; fp.gidx = gidx;
  fp.oxyz = out;                 // 512*14*3
  fp.oR   = out + 21504;         // 512*9
  fp.ot   = out + 26112;         // 512*3
  fp.os   = out + 27648;         // 512*16
  final_kernel<<<LSEQ, 128, 0, stream>>>(fp);
}
